// TimeCondAttention_74191265071598
// MI455X (gfx1250) — matrix operand order repaired and checked by reading, not compiled, not run on hardware
//
#include <hip/hip_runtime.h>
#include <hip/hip_bf16.h>
#include <stdint.h>

typedef __bf16 bf16;
typedef __attribute__((ext_vector_type(16))) __bf16 v16bf;
typedef __attribute__((ext_vector_type(8)))  __bf16 v8bf;
typedef __attribute__((ext_vector_type(8)))  float  v8f;

#define B_  4
#define N_  1024
#define D_  512
#define H_  8
#define DH_ 64
#define TC_ 512
#define AB_ 16
#define NN_ (N_ * N_)
#define NEG_ 1e6f

// ---------------------------------------------------------------------------
// WMMA helper: D = A(16x32 bf16) * B(32x16 bf16) + C(16x16 f32)
// ---------------------------------------------------------------------------
__device__ __forceinline__ v8f wmma_bf(v16bf a, v16bf b, v8f c) {
  return __builtin_amdgcn_wmma_f32_16x16x32_bf16(false, a, false, b, (short)0, c,
                                                 false, false);
}

// A-matrix 16x32 bf16 fragment for one lane.
// Lane (group g = lane>>4) element e maps to K = e + 16*g.
__device__ __forceinline__ v16bf load_a_frag(const bf16* row, int g) {
  v8bf lo = *(const v8bf*)(row + 16 * g);
  v8bf hi = *(const v8bf*)(row + 16 * g + 8);
  v16bf a;
#pragma unroll
  for (int i = 0; i < 8; ++i) { a[i] = lo[i]; a[8 + i] = hi[i]; }
  return a;
}

// ---------------------------------------------------------------------------
// K0: f32 -> bf16 convert (for weight matrices)
// ---------------------------------------------------------------------------
__global__ __launch_bounds__(256) void cvt_bf16_kernel(const float* __restrict__ src,
                                                       bf16* __restrict__ dst, int n) {
  int i = blockIdx.x * 256 + threadIdx.x;
  if (i < n) dst[i] = (bf16)src[i];
}

// ---------------------------------------------------------------------------
// K1: FiLM  t[b, col] = sum_tc silu(time[b,tc]) * Wt[col,tc] + bt[col]
// ---------------------------------------------------------------------------
__global__ __launch_bounds__(256) void film_kernel(const float* __restrict__ time,
                                                   const float* __restrict__ Wt,
                                                   const float* __restrict__ bt,
                                                   float* __restrict__ ss) {
  int gid = blockIdx.x * 256 + threadIdx.x;        // B * 2D = 4096
  int b = gid >> 10;
  int col = gid & 1023;
  const float* tp = time + (size_t)b * TC_;
  const float* wr = Wt + (size_t)col * TC_;
  float acc = bt[col];
  for (int k = 0; k < TC_; ++k) {
    float t = tp[k];
    float s = t / (1.f + __expf(-t));              // silu
    acc += s * wr[k];
  }
  ss[gid] = acc;
}

// ---------------------------------------------------------------------------
// K2: LayerNorm + FiLM + mask -> xn (bf16);  also x -> bf16 (raw context)
// ---------------------------------------------------------------------------
__global__ __launch_bounds__(256) void ln_film_kernel(const float* __restrict__ x,
                                                      const float* __restrict__ ss,
                                                      const float* __restrict__ gamma,
                                                      const float* __restrict__ seq_mask,
                                                      bf16* __restrict__ xnbf,
                                                      bf16* __restrict__ xbf) {
  int row = blockIdx.x;                            // b*N + n
  int b = row >> 10;
  const float* xr = x + (size_t)row * D_;
  float vals[2], s1 = 0.f, s2 = 0.f;
#pragma unroll
  for (int i = 0; i < 2; ++i) {
    float v = xr[threadIdx.x + 256 * i];
    vals[i] = v; s1 += v; s2 += v * v;
  }
#pragma unroll
  for (int off = 1; off < 32; off <<= 1) {
    s1 += __shfl_xor(s1, off, 32);
    s2 += __shfl_xor(s2, off, 32);
  }
  __shared__ float sh1[8], sh2[8];
  int w = threadIdx.x >> 5;
  if ((threadIdx.x & 31) == 0) { sh1[w] = s1; sh2[w] = s2; }
  __syncthreads();
  s1 = 0.f; s2 = 0.f;
#pragma unroll
  for (int i = 0; i < 8; ++i) { s1 += sh1[i]; s2 += sh2[i]; }
  float mu = s1 * (1.f / D_);
  float var = s2 * (1.f / D_) - mu * mu;
  float rs = rsqrtf(var + 1e-5f);
  float mask = seq_mask[row];
  const float* sc = ss + (size_t)b * (2 * D_);
#pragma unroll
  for (int i = 0; i < 2; ++i) {
    int d = threadIdx.x + 256 * i;
    float xn = (vals[i] - mu) * rs * gamma[d];
    xn = xn * (sc[d] + 1.f) + sc[D_ + d];
    xn *= mask;
    xnbf[(size_t)row * D_ + d] = (bf16)xn;
    xbf[(size_t)row * D_ + d] = (bf16)vals[i];
  }
}

// ---------------------------------------------------------------------------
// K3/K6: WMMA GEMM  C[Mtot,Nout] = A[Mtot,K] * W[Nout,K]^T
// one wave per 16x64 C tile: one A fragment feeds 4 WMMAs per k-step.
//   mode 0: q  -> bf16 [B,H,N,DH], scaled by DH^-0.5
//   mode 1: kv -> k bf16 [B,H,N,DH] (cols<512), vT bf16 [B,H,DH,N] (cols>=512)
//   mode 2: out -> f32 [B,N,D] * seq_mask
// ---------------------------------------------------------------------------
__global__ __launch_bounds__(256) void gemm_wmma_kernel(const bf16* __restrict__ A,
                                                        const bf16* __restrict__ W,
                                                        int Nout, int mode,
                                                        const float* __restrict__ seq_mask,
                                                        bf16* __restrict__ oq,
                                                        bf16* __restrict__ ok,
                                                        bf16* __restrict__ ovT,
                                                        float* __restrict__ of32) {
  const int K = D_;
  int wave = (blockIdx.x * 256 + threadIdx.x) >> 5;
  int lane = threadIdx.x & 31;
  int g = lane >> 4, lc = lane & 15;
  int ntn = Nout >> 6;                                 // 64-wide N tiles
  int tm = wave / ntn, tn = wave % ntn;

  const bf16* arow = A + (size_t)(tm * 16 + lc) * K;   // A frag: lane row
  const bf16* wrow = W + (size_t)(tn * 64 + lc) * K;   // B frag: lane col (sub-tile +16*t rows)
  v8f acc[4];
#pragma unroll
  for (int t = 0; t < 4; ++t) acc[t] = (v8f)(0.0f);
#pragma unroll 2
  for (int k0 = 0; k0 < K; k0 += 32) {
    v16bf af = load_a_frag(arow + k0, g);
#pragma unroll
    for (int t = 0; t < 4; ++t) {
      v16bf bfg = *(const v16bf*)(wrow + (size_t)(t * 16) * K + k0 + 16 * g);
      acc[t] = wmma_bf(af, bfg, acc[t]);
    }
  }
#pragma unroll
  for (int t = 0; t < 4; ++t) {
#pragma unroll
    for (int r = 0; r < 8; ++r) {
      int m = tm * 16 + r + 8 * g;                     // global row (b*N+n)
      int b = m >> 10, n = m & 1023;
      int c = tn * 64 + t * 16 + lc;
      float v = acc[t][r];
      if (mode == 0) {
        int h = c >> 6, d = c & 63;
        oq[(((size_t)(b * H_ + h) << 10) + n) * DH_ + d] = (bf16)(v * 0.125f);
      } else if (mode == 1) {
        if (c < H_ * DH_) {
          int h = c >> 6, d = c & 63;
          ok[(((size_t)(b * H_ + h) << 10) + n) * DH_ + d] = (bf16)v;
        } else {
          int c2 = c - H_ * DH_;
          int h = c2 >> 6, d = c2 & 63;
          ovT[((size_t)(b * H_ + h) * DH_ + d) * N_ + n] = (bf16)v;
        }
      } else {
        of32[(size_t)m * D_ + c] = v * seq_mask[m];
      }
    }
  }
}

// ---------------------------------------------------------------------------
// K4: bias projection  bias[b,h,p] = sum_a attn_bias[b,a,p] * Wb[h,a] + bb[h]
// One wave -> 16 flat (i,j)-positions for ALL 8 heads with one WMMA.
// attn_bias (256 MB, the roofline term) is read exactly once.
// ---------------------------------------------------------------------------
__global__ __launch_bounds__(256) void bias_wmma_kernel(const float* __restrict__ attn_bias,
                                                        const float* __restrict__ Wb,
                                                        const float* __restrict__ bb,
                                                        bf16* __restrict__ biasbf) {
  int wave = (blockIdx.x * 256 + threadIdx.x) >> 5;
  int lane = threadIdx.x & 31;
  int g = lane >> 4, lc = lane & 15;
  const int tilesPerBatch = NN_ / 16;
  int b = wave / tilesPerBatch;
  int base = (wave % tilesPerBatch) * 16;

  const float* src = attn_bias + (size_t)b * AB_ * NN_ + base + lc;
  v16bf af;
#pragma unroll
  for (int e = 0; e < 8; ++e) {
    af[e] = (bf16)src[(size_t)(e + 8 * g) * NN_];      // a = e + 8g (0..15)
    af[8 + e] = (bf16)0.f;                             // K pad
  }
  v16bf bfg;
#pragma unroll
  for (int e = 0; e < 16; ++e) {
    int a = (e & 7) + 8 * g + 16 * (e >> 3);
    float wv = (lc < H_ && a < AB_) ? Wb[lc * AB_ + a] : 0.f;
    bfg[e] = (bf16)wv;
  }
  float cb = (lc < H_) ? bb[lc] : 0.f;
  v8f c;
#pragma unroll
  for (int r = 0; r < 8; ++r) c[r] = cb;
  c = wmma_bf(af, bfg, c);
  if (lc < H_) {
    size_t ob = ((size_t)b * H_ + lc) * NN_ + base;
#pragma unroll
    for (int r = 0; r < 8; ++r) biasbf[ob + r + 8 * g] = (bf16)c[r];
  }
}

// ---------------------------------------------------------------------------
// Async copy of one 16x32 bf16 bias tile (1 KB) into LDS:
// two global_load_async_to_lds_b128 ops (32 lanes x 16B each), ASYNCcnt-tracked.
// ---------------------------------------------------------------------------
__device__ __forceinline__ void async_bias_tile(const bf16* __restrict__ bbias,
                                                int i0, int j0, bf16* dst, int lane) {
  int row = lane >> 2;                                  // 0..7
  int quad = lane & 3;                                  // 16B sub-chunk in a 64B row
  const bf16* g0 = bbias + ((size_t)(i0 + row) * N_ + j0 + quad * 8);
  const bf16* g1 = bbias + ((size_t)(i0 + 8 + row) * N_ + j0 + quad * 8);
  uint32_t l0 = (uint32_t)(uintptr_t)(dst + row * 32 + quad * 8);
  uint32_t l1 = (uint32_t)(uintptr_t)(dst + (8 + row) * 32 + quad * 8);
  asm volatile("global_load_async_to_lds_b128 %0, %1, off\n\t"
               "global_load_async_to_lds_b128 %2, %3, off"
               :
               : "v"(l0), "v"(g0), "v"(l1), "v"(g1)
               : "memory");
}

// ---------------------------------------------------------------------------
// K5: flash attention. One wave per (b, h, 16-row i-tile); j in steps of 32.
// Bias tiles are double-buffered through LDS with async-to-LDS copies so the
// only HBM stream in the loop is hidden behind the WMMA/softmax work.
// ---------------------------------------------------------------------------
__global__ __launch_bounds__(128) void flash_attn_kernel(const bf16* __restrict__ qb,
                                                         const bf16* __restrict__ kb,
                                                         const bf16* __restrict__ vT,
                                                         const bf16* __restrict__ biasbf,
                                                         const float* __restrict__ seq_mask,
                                                         bf16* __restrict__ obf) {
  __shared__ bf16 pbuf[4][16 * 32];       // P transpose staging (per wave)
  __shared__ bf16 bstage[4][2][16 * 32];  // bias double-buffer (per wave)
  int w = threadIdx.x >> 5, lane = threadIdx.x & 31;
  int g = lane >> 4, lc = lane & 15;
  int tile = blockIdx.x * 4 + w;                       // 0..2047
  int itile = tile & 63;
  int bh = tile >> 6;                                  // b*H + h
  int b = bh >> 3, h = bh & 7;
  int i0 = itile * 16;

  const bf16* qrow = qb + ((size_t)bh * N_ + i0 + lc) * DH_;
  v16bf qf0 = load_a_frag(qrow, g);                    // d = 0..31
  v16bf qf1 = load_a_frag(qrow + 32, g);               // d = 32..63
  const bf16* kbase = kb + (size_t)bh * N_ * DH_;
  const bf16* vbase = vT + (size_t)bh * DH_ * N_;
  const bf16* bbias = biasbf + (size_t)bh * NN_;
  const float* smb = seq_mask + (size_t)b * N_;

  float mrow[8], mst[8], lst[8];
#pragma unroll
  for (int r = 0; r < 8; ++r) {
    mrow[r] = smb[i0 + r + 8 * g];
    mst[r] = -1e30f;
    lst[r] = 0.f;
  }
  v8f oacc[4];
#pragma unroll
  for (int t = 0; t < 4; ++t) oacc[t] = (v8f)(0.0f);

  // prologue: stage first bias tile
  async_bias_tile(bbias, i0, 0, &bstage[w][0][0], lane);

  for (int it = 0; it < N_ / 32; ++it) {
    int j0 = it * 32;
    // prefetch next bias tile into the other buffer, then wait for current
    if (it + 1 < N_ / 32) {
      async_bias_tile(bbias, i0, j0 + 32, &bstage[w][(it + 1) & 1][0], lane);
      asm volatile("s_wait_asynccnt 0x2" ::: "memory");
    } else {
      asm volatile("s_wait_asynccnt 0x0" ::: "memory");
    }
    const bf16* bs = &bstage[w][it & 1][0];
    v8f c0, c1;
#pragma unroll
    for (int r = 0; r < 8; ++r) {
      int rr = (r + 8 * g) * 32;
      c0[r] = (float)bs[rr + lc];
      c1[r] = (float)bs[rr + 16 + lc];
    }
    // K fragments (B = k^T: lane col = j, K = d)
    const bf16* kr0 = kbase + (size_t)(j0 + lc) * DH_;
    const bf16* kr1 = kbase + (size_t)(j0 + 16 + lc) * DH_;
    v16bf k00 = *(const v16bf*)(kr0 + 16 * g);
    v16bf k01 = *(const v16bf*)(kr0 + 32 + 16 * g);
    v16bf k10 = *(const v16bf*)(kr1 + 16 * g);
    v16bf k11 = *(const v16bf*)(kr1 + 32 + 16 * g);
    c0 = wmma_bf(qf0, k00, c0);
    c0 = wmma_bf(qf1, k01, c0);
    c1 = wmma_bf(qf0, k10, c1);
    c1 = wmma_bf(qf1, k11, c1);

    float mc0 = smb[j0 + lc], mc1 = smb[j0 + 16 + lc];
#pragma unroll
    for (int r = 0; r < 8; ++r) {
      c0[r] -= (1.f - mrow[r] * mc0) * NEG_;
      c1[r] -= (1.f - mrow[r] * mc1) * NEG_;
    }
    // online softmax
    float corr[8];
#pragma unroll
    for (int r = 0; r < 8; ++r) {
      float v = fmaxf(c0[r], c1[r]);
#pragma unroll
      for (int off = 1; off < 16; off <<= 1) v = fmaxf(v, __shfl_xor(v, off, 32));
      float nm = fmaxf(mst[r], v);
      corr[r] = __expf(mst[r] - nm);
      mst[r] = nm;
      c0[r] = __expf(c0[r] - nm);
      c1[r] = __expf(c1[r] - nm);
      float s = c0[r] + c1[r];
#pragma unroll
      for (int off = 1; off < 16; off <<= 1) s += __shfl_xor(s, off, 32);
      lst[r] = lst[r] * corr[r] + s;
    }
#pragma unroll
    for (int t = 0; t < 4; ++t)
#pragma unroll
      for (int r = 0; r < 8; ++r) oacc[t][r] *= corr[r];

    // P: C layout -> LDS (row-major 16x32) -> A fragment
#pragma unroll
    for (int r = 0; r < 8; ++r) {
      int row = r + 8 * g;
      pbuf[w][row * 32 + lc] = (bf16)c0[r];
      pbuf[w][row * 32 + 16 + lc] = (bf16)c1[r];
    }
    v16bf pf = load_a_frag(&pbuf[w][lc * 32], g);

    // PV: B fragments from vT are contiguous (K = j'' = e + 16g)
#pragma unroll
    for (int t = 0; t < 4; ++t) {
      v16bf vf = *(const v16bf*)(vbase + (size_t)(16 * t + lc) * N_ + j0 + 16 * g);
      oacc[t] = wmma_bf(pf, vf, oacc[t]);
    }
  }
  // epilogue: normalize and store bf16 [B, N, H*DH]
#pragma unroll
  for (int t = 0; t < 4; ++t)
#pragma unroll
    for (int r = 0; r < 8; ++r) {
      int n = i0 + r + 8 * g;
      obf[((size_t)b * N_ + n) * (H_ * DH_) + h * DH_ + 16 * t + lc] =
          (bf16)(oacc[t][r] / lst[r]);
    }
}

// ---------------------------------------------------------------------------
// host launcher
// ---------------------------------------------------------------------------
extern "C" void kernel_launch(void* const* d_in, const int* in_sizes, int n_in,
                              void* d_out, int out_size, void* d_ws, size_t ws_size,
                              hipStream_t stream) {
  const float* x        = (const float*)d_in[0];
  const float* time     = (const float*)d_in[1];
  const float* attnbias = (const float*)d_in[2];
  const float* seq_mask = (const float*)d_in[3];
  const float* gamma    = (const float*)d_in[4];
  const float* Wt       = (const float*)d_in[5];
  const float* bt       = (const float*)d_in[6];
  const float* Wq       = (const float*)d_in[7];
  const float* Wkv      = (const float*)d_in[8];
  const float* Wo       = (const float*)d_in[9];
  const float* Wb       = (const float*)d_in[10];
  const float* bb       = (const float*)d_in[11];
  float* out = (float*)d_out;

  char* ws = (char*)d_ws;
  size_t off = 0;
  auto alloc = [&](size_t bytes) -> char* {
    char* p = ws + off;
    off = (off + bytes + 255) & ~(size_t)255;
    return p;
  };
  float* ss    = (float*)alloc((size_t)B_ * 2 * D_ * 4);
  bf16* xbf    = (bf16*)alloc((size_t)B_ * N_ * D_ * 2);
  bf16* xnbf   = (bf16*)alloc((size_t)B_ * N_ * D_ * 2);
  bf16* wqbf   = (bf16*)alloc((size_t)H_ * DH_ * D_ * 2);
  bf16* wkvbf  = (bf16*)alloc((size_t)2 * H_ * DH_ * D_ * 2);
  bf16* wobf   = (bf16*)alloc((size_t)D_ * H_ * DH_ * 2);
  bf16* qbf    = (bf16*)alloc((size_t)B_ * H_ * N_ * DH_ * 2);
  bf16* kbf    = (bf16*)alloc((size_t)B_ * H_ * N_ * DH_ * 2);
  bf16* vTbf   = (bf16*)alloc((size_t)B_ * H_ * DH_ * N_ * 2);
  bf16* obf    = (bf16*)alloc((size_t)B_ * N_ * H_ * DH_ * 2);
  bf16* biasbf = (bf16*)alloc((size_t)B_ * H_ * NN_ * 2);
  (void)ws_size; (void)in_sizes; (void)n_in; (void)out_size;

  cvt_bf16_kernel<<<(H_ * DH_ * D_ + 255) / 256, 256, 0, stream>>>(Wq, wqbf, H_ * DH_ * D_);
  cvt_bf16_kernel<<<(2 * H_ * DH_ * D_ + 255) / 256, 256, 0, stream>>>(Wkv, wkvbf, 2 * H_ * DH_ * D_);
  cvt_bf16_kernel<<<(D_ * H_ * DH_ + 255) / 256, 256, 0, stream>>>(Wo, wobf, D_ * H_ * DH_);

  film_kernel<<<(B_ * 2 * D_) / 256, 256, 0, stream>>>(time, Wt, bt, ss);

  ln_film_kernel<<<B_ * N_, 256, 0, stream>>>(x, ss, gamma, seq_mask, xnbf, xbf);

  // q = xn @ Wq^T : 256 Mtiles x 8 Ntiles(64) = 2048 waves -> 256 blocks
  gemm_wmma_kernel<<<((B_ * N_ / 16) * (H_ * DH_ / 64)) / 8, 256, 0, stream>>>(
      xnbf, wqbf, H_ * DH_, 0, seq_mask, qbf, nullptr, nullptr, nullptr);
  // k,v = x @ Wkv^T : 256 x 16 = 4096 waves -> 512 blocks
  gemm_wmma_kernel<<<((B_ * N_ / 16) * (2 * H_ * DH_ / 64)) / 8, 256, 0, stream>>>(
      xbf, wkvbf, 2 * H_ * DH_, 1, seq_mask, nullptr, kbf, vTbf, nullptr);

  bias_wmma_kernel<<<(B_ * NN_ / 16) / 8, 256, 0, stream>>>(attnbias, Wb, bb, biasbf);

  flash_attn_kernel<<<(B_ * H_ * (N_ / 16)) / 4, 128, 0, stream>>>(
      qbf, kbf, vTbf, biasbf, seq_mask, obf);

  // out = O @ Wo^T : 2048 waves -> 256 blocks
  gemm_wmma_kernel<<<((B_ * N_ / 16) * (D_ / 64)) / 8, 256, 0, stream>>>(
      obf, wobf, D_, 2, seq_mask, nullptr, nullptr, nullptr, out);
}
